// WeightConv_50577534877938
// MI455X (gfx1250) — compile-verified, matrix operand order checked by reading, not run on hardware
//
#include <hip/hip_runtime.h>

typedef __attribute__((ext_vector_type(16))) __bf16 v16bf;
typedef __attribute__((ext_vector_type(8)))  float  v8f;

#define Bn 16
#define Hh 128
#define Ww 128

// ---------- helpers ----------
static __device__ __forceinline__ unsigned short f2bf(float f) {
  unsigned u = __float_as_uint(f);
  u += 0x7FFFu + ((u >> 16) & 1u);          // round-to-nearest-even
  return (unsigned short)(u >> 16);
}
static __device__ __forceinline__ unsigned pack2(float a, float b) {
  return (unsigned)f2bf(a) | ((unsigned)f2bf(b) << 16);
}

// ---------- kernel 1: global average pool -> s[b*64+c] ----------
__global__ __launch_bounds__(256) void pool_kernel(const float* __restrict__ x,
                                                   float* __restrict__ s) {
  __shared__ float red[256];
  const int bc = blockIdx.x;                 // 0..1023 = b*64+c
  const int t  = threadIdx.x;
  const float4* p4 = (const float4*)(x + (size_t)bc * (Hh * Ww));
  float sum = 0.f;
#pragma unroll
  for (int k = 0; k < 16; ++k) {
    float4 v = p4[k * 256 + t];
    sum += v.x + v.y + v.z + v.w;
  }
  red[t] = sum;
  __syncthreads();
  for (int off = 128; off > 0; off >>= 1) {
    if (t < off) red[t] += red[t + off];
    __syncthreads();
  }
  if (t == 0) s[bc] = red[0] * (1.0f / 16384.0f);
}

// ---------- kernel 2: MLP + dynamic weights + WMMA conv ----------
#define WLDS_BYTES 73728                     // 4 mt * 18 ks * 32 lanes * 32B (A-frags)
#define XROWS 6
#define XCOLS 130
#define PIXD  36                             // dwords per pixel (72 bf16; 9 uint4)
#define PIXQ  9                              // uint4 per pixel
#define XLDS_DW (XROWS * XCOLS * PIXD)
#define XLDS_BYTES (XLDS_DW * 4)             // 112320
#define H2_OFF (WLDS_BYTES + XLDS_BYTES)     // 186048
#define H1_OFF (H2_OFF + 2048 * 4)           // 194240
#define SMEM_BYTES (H1_OFF + 64 * 4)         // 194496

__global__ __launch_bounds__(256, 1) void wconv_kernel(
    const float* __restrict__ x,  const float* __restrict__ s,
    const float* __restrict__ w1, const float* __restrict__ b1,
    const float* __restrict__ w2, const float* __restrict__ b2,
    const float* __restrict__ w3, float* __restrict__ out) {
  extern __shared__ __align__(128) unsigned char smem[];
  unsigned* wlds_u = (unsigned*)smem;                      // A-fragments (bf16 pairs)
  unsigned* xlds_u = (unsigned*)(smem + WLDS_BYTES);       // input tile (bf16 pairs)
  const uint4* xlds_q = (const uint4*)(smem + WLDS_BYTES); // same, 16B granules
  float* h2s = (float*)(smem + H2_OFF);                    // 16 x 128
  float* h1s = (float*)(smem + H1_OFF);                    // 16 x 4

  const int tid  = threadIdx.x;
  const int lane = tid & 31;
  const int wid  = tid >> 5;                 // 8 waves
  const int b    = blockIdx.x;               // batch
  const int h0   = blockIdx.y * 4;           // output rows h0..h0+3

  // --- fc1: h1[b'][j] = s[b'] @ w1[j] + b1[j] ---
  if (tid < 64) {
    const int bb = tid >> 2, j = tid & 3;
    float acc = b1[j];
    const float* sp = s  + bb * 64;
    const float* wp = w1 + j * 64;
#pragma unroll 8
    for (int c = 0; c < 64; ++c) acc += sp[c] * wp[c];
    h1s[tid] = acc;
  }
  __syncthreads();

  // --- fc2 + sigmoid: h2[b'][g] ---
  for (int i = tid; i < 2048; i += 256) {
    const int bb = i >> 7, g = i & 127;
    const float* hp = h1s + bb * 4;
    const float* wp = w2 + g * 4;
    float v = b2[g] + hp[0]*wp[0] + hp[1]*wp[1] + hp[2]*wp[2] + hp[3]*wp[3];
    h2s[i] = 1.0f / (1.0f + __expf(-v));
  }
  __syncthreads();

  // --- dynamic weights for this batch, written in WMMA A-fragment order ---
  // k-order: k = (kh*3+kw)*64 + c ; fragment: [mt][ks][lane][8 dwords]
  for (int f = tid; f < 4 * 18 * 32 * 8; f += 256) {
    const int d  = f & 7;                    // dword (VGPR) within fragment
    const int ln = (f >> 3) & 31;
    const int r  = f >> 8;
    const int ks = r % 18;
    const int mt = r / 18;
    const int o  = mt * 16 + (ln & 15);
    const int hi = ln >> 4;
    const int kkidx = ks >> 1;               // 0..8 tap
    const int c0 = (ks & 1) * 32 + (d >> 2) * 16 + hi * 8 + (d & 3) * 2;
    const int osel = o >> 5;
    const int g0 = (c0 & 3) * 32 + b * 2 + osel;
    const int g1 = ((c0 + 1) & 3) * 32 + b * 2 + osel;
    const int widx = (o & 31) * 9 + kkidx;
    const float v0 = h2s[(c0 >> 2) * 128 + g0] * w3[g0 * 288 + widx];
    const float v1 = h2s[(c0 >> 2) * 128 + g1] * w3[g1 * 288 + widx];
    wlds_u[f] = pack2(v0, v1);
  }

  // --- stage input halo tile: rows h0-1..h0+4, cols -1..128, bf16, c-fastest ---
  for (int f = tid; f < XROWS * XCOLS * 32; f += 256) {
    const int col = f % XCOLS;
    const int r2  = f / XCOLS;
    const int cp  = r2 & 31;                 // channel pair
    const int row = r2 >> 5;
    const int gh = h0 - 1 + row;
    const int gw = col - 1;
    float v0 = 0.f, v1 = 0.f;
    if (gh >= 0 && gh < Hh && gw >= 0 && gw < Ww) {
      const float* px = x + (((size_t)b * 64 + cp * 2) * Hh + gh) * Ww + gw;
      v0 = px[0];
      v1 = px[(size_t)Hh * Ww];
    }
    xlds_u[(row * XCOLS + col) * PIXD + cp] = pack2(v0, v1);
  }
  __syncthreads();

  // --- WMMA main loop: wave tile = 2 M-tiles x (2x16 cols) x 4 rows ---
  const int mtg = wid & 1;                   // M pair: mt = 2*mtg + {0,1}
  const int ntg = wid >> 1;                  // 0..3: cols ntg*32 + {0,16}
  const int colbase = ntg * 32 + (lane & 15);
  const int hiq = lane >> 4;                 // 0/1: hi half-wave
  const int hi8 = hiq << 3;                  // +8 K / +8 M for hi half-wave

  v8f acc[4][4] = {};                        // [hr][mti*2+nti]

  for (int ks = 0; ks < 18; ++ks) {
    v16bf a0 = *(const v16bf*)(smem + ((((mtg * 2 + 0) * 18 + ks) * 32 + lane) << 5));
    v16bf a1 = *(const v16bf*)(smem + ((((mtg * 2 + 1) * 18 + ks) * 32 + lane) << 5));
    const int kkidx = ks >> 1;
    const int dh = kkidx / 3;
    const int dw = kkidx - dh * 3;
    const int q4 = (ks & 1) * 4 + hiq;       // uint4 offset within pixel
#pragma unroll
    for (int hr = 0; hr < 4; ++hr) {
      const uint4* pb = xlds_q + ((hr + dh) * XCOLS + colbase + dw) * PIXQ + q4;
      union { uint4 q[2]; v16bf v; } b0, b1;
      b0.q[0] = pb[0];              b0.q[1] = pb[2];               // cols +0
      b1.q[0] = pb[16 * PIXQ];      b1.q[1] = pb[16 * PIXQ + 2];   // cols +16
      acc[hr][0] = __builtin_amdgcn_wmma_f32_16x16x32_bf16(
          false, a0, false, b0.v, (short)0, acc[hr][0], false, false);
      acc[hr][1] = __builtin_amdgcn_wmma_f32_16x16x32_bf16(
          false, a0, false, b1.v, (short)0, acc[hr][1], false, false);
      acc[hr][2] = __builtin_amdgcn_wmma_f32_16x16x32_bf16(
          false, a1, false, b0.v, (short)0, acc[hr][2], false, false);
      acc[hr][3] = __builtin_amdgcn_wmma_f32_16x16x32_bf16(
          false, a1, false, b1.v, (short)0, acc[hr][3], false, false);
    }
  }

  // --- store D tiles: lane = N col, VGPR j = M row (+8 for hi half-wave) ---
#pragma unroll
  for (int hr = 0; hr < 4; ++hr) {
#pragma unroll
    for (int mti = 0; mti < 2; ++mti) {
#pragma unroll
      for (int nti = 0; nti < 2; ++nti) {
        const int orow = (mtg * 2 + mti) * 16 + hi8;
        float* ob = out + (((size_t)b * 64 + orow) * Hh + (h0 + hr)) * Ww
                        + colbase + nti * 16;
        const v8f& a = acc[hr][mti * 2 + nti];
#pragma unroll
        for (int j = 0; j < 8; ++j)
          ob[(size_t)j * Hh * Ww] = a[j];
      }
    }
  }
}

// ---------- launch ----------
extern "C" void kernel_launch(void* const* d_in, const int* in_sizes, int n_in,
                              void* d_out, int out_size, void* d_ws, size_t ws_size,
                              hipStream_t stream) {
  const float* x  = (const float*)d_in[0];
  const float* w1 = (const float*)d_in[1];
  const float* b1 = (const float*)d_in[2];
  const float* w2 = (const float*)d_in[3];
  const float* b2 = (const float*)d_in[4];
  const float* w3 = (const float*)d_in[5];
  float* out  = (float*)d_out;
  float* sbuf = (float*)d_ws;               // 1024 floats

  (void)in_sizes; (void)n_in; (void)out_size; (void)ws_size;
  hipFuncSetAttribute((const void*)wconv_kernel,
                      hipFuncAttributeMaxDynamicSharedMemorySize, SMEM_BYTES);

  pool_kernel<<<dim3(Bn * 64), dim3(256), 0, stream>>>(x, sbuf);
  wconv_kernel<<<dim3(Bn, Hh / 4), dim3(256), SMEM_BYTES, stream>>>(
      x, sbuf, w1, b1, w2, b2, w3, out);
}
